// CPLinear_60361470378143
// MI455X (gfx1250) — matrix operand order repaired and checked by reading, not compiled, not run on hardware
//
#include <hip/hip_runtime.h>
#include <math.h>

// ---------------------------------------------------------------------------
// CPLinear for MI455X (gfx1250): fused low-rank QKV projection.
//   Stage 0: one-time split of x (and transposed W) into bf16 hi/lo pairs.
//            (hi+lo carries ~16 mantissa bits -> fp32-like GEMM accuracy.)
//   Stage 1: six GEMMs  Y = x @ W  via V_WMMA_F32_16X16X32_BF16.
//            K-contiguous bf16 rows mean A/B WMMA fragments are loaded
//            straight from global memory as 16B chunks in a contiguous-k (k = 16*half + e, same for A and B)
//            layout: no LDS, no barriers, no in-loop conversion.
//   Stage 2: per-token rank contraction + RoPE (RoPE commutes with the
//            rank contraction since it is elementwise-linear in head_dim).
// ---------------------------------------------------------------------------

typedef __attribute__((ext_vector_type(16))) __bf16 v16bf;
typedef __attribute__((ext_vector_type(8)))  __bf16 v8bf;
typedef __attribute__((ext_vector_type(4)))  __bf16 v4bf;
typedef __attribute__((ext_vector_type(8)))  float  v8f;

namespace {
constexpr int D_IN   = 2048;   // K of every GEMM
constexpr int BSROWS = 16384;  // B*S rows
constexpr int SEQ    = 4096;   // S (for RoPE position)
constexpr int NHEAD  = 16;
constexpr int HDIM   = 128;
constexpr int QRANK  = 8;
constexpr int KVRANK = 2;
constexpr int TILE_M = 128;    // rows per workgroup (8 waves x 16)
constexpr int KSTEP  = 32;     // K per WMMA
}

// ---------------------------------------------------------------------------
// Stage 0a: elementwise split  f32 -> bf16 hi + bf16 lo  (same layout).
// One thread per 4 elements.
// ---------------------------------------------------------------------------
__global__ __launch_bounds__(256)
void cpl_split(const float* __restrict__ in, __bf16* __restrict__ hi,
               __bf16* __restrict__ lo, int n4) {
  int i = blockIdx.x * 256 + threadIdx.x;
  if (i >= n4) return;
  float4 v = ((const float4*)in)[i];
  v4bf h, l;
  h.x = (__bf16)v.x;  l.x = (__bf16)(v.x - (float)h.x);
  h.y = (__bf16)v.y;  l.y = (__bf16)(v.y - (float)h.y);
  h.z = (__bf16)v.z;  l.z = (__bf16)(v.z - (float)h.z);
  h.w = (__bf16)v.w;  l.w = (__bf16)(v.w - (float)h.w);
  ((v4bf*)hi)[i] = h;
  ((v4bf*)lo)[i] = l;
}

// ---------------------------------------------------------------------------
// Stage 0b: split + transpose  W[k][n] (f32) -> Wt[n][k] (bf16 hi/lo).
// One thread per element, k fastest (coalesced writes; W is tiny/L2).
// ---------------------------------------------------------------------------
__global__ __launch_bounds__(256)
void cpl_split_transpose(const float* __restrict__ W, __bf16* __restrict__ hiT,
                         __bf16* __restrict__ loT, int N) {
  size_t idx = (size_t)blockIdx.x * 256 + threadIdx.x;   // 0 .. N*2048-1
  int k = (int)(idx & (D_IN - 1));
  int n = (int)(idx >> 11);
  float v = W[(size_t)k * N + n];
  __bf16 h = (__bf16)v;
  hiT[idx] = h;
  loT[idx] = (__bf16)(v - (float)h);
}

// ---------------------------------------------------------------------------
// Stage 1 GEMM: Y[M x N] = X[M x 2048] @ W[2048 x N], split-bf16 WMMA.
// grid = (M/128, N/(NT*16)), block = 256 (8 waves).
// Each wave: 16 rows x NT 16-col C tiles; all fragments direct from global.
//   A (16x32 bf16): lane(m=l16, half) elems e <-> K=half*16+e
//                   -> two b128 loads from row m.
//   B (32x16 bf16): lane(n=l16, half) elems e <-> K=half*16+e
//                   -> two b128 loads from Wt row n.
// ---------------------------------------------------------------------------
template <int NT>
__global__ __launch_bounds__(256)
void cpl_gemm_wmma(const __bf16* __restrict__ Xhi, const __bf16* __restrict__ Xlo,
                   const __bf16* __restrict__ Wthi, const __bf16* __restrict__ Wtlo,
                   float* __restrict__ Y, int N) {
  const int tid  = threadIdx.x;
  const int wave = tid >> 5;              // 0..7  (wave32)
  const int lane = tid & 31;
  const int half = lane >> 4;             // 0/1  (lane group)
  const int l16  = lane & 15;

  const int row0 = blockIdx.x * TILE_M;
  const int n0   = blockIdx.y * (NT * 16);

  const int m = row0 + wave * 16 + l16;   // A row owned by this lane
  const __bf16* xh = Xhi + (size_t)m * D_IN;
  const __bf16* xl = Xlo + (size_t)m * D_IN;

  const __bf16* bh[NT];
  const __bf16* bl[NT];
#pragma unroll
  for (int t = 0; t < NT; ++t) {
    bh[t] = Wthi + (size_t)(n0 + t * 16 + l16) * D_IN;
    bl[t] = Wtlo + (size_t)(n0 + t * 16 + l16) * D_IN;
  }

  v8f acc[NT];
#pragma unroll
  for (int t = 0; t < NT; ++t) acc[t] = (v8f){};

#pragma unroll 4
  for (int k0 = 0; k0 < D_IN; k0 += KSTEP) {
    v16bf ahi, alo;
    ((v8bf*)&ahi)[0] = *(const v8bf*)(xh + k0 + half * 16);
    ((v8bf*)&ahi)[1] = *(const v8bf*)(xh + k0 + half * 16 + 8);
    ((v8bf*)&alo)[0] = *(const v8bf*)(xl + k0 + half * 16);
    ((v8bf*)&alo)[1] = *(const v8bf*)(xl + k0 + half * 16 + 8);

#pragma unroll
    for (int t = 0; t < NT; ++t) {
      v16bf bhi, blo;
      ((v8bf*)&bhi)[0] = *(const v8bf*)(bh[t] + k0 + half * 16);
      ((v8bf*)&bhi)[1] = *(const v8bf*)(bh[t] + k0 + half * 16 + 8);
      ((v8bf*)&blo)[0] = *(const v8bf*)(bl[t] + k0 + half * 16);
      ((v8bf*)&blo)[1] = *(const v8bf*)(bl[t] + k0 + half * 16 + 8);

      // split-bf16: hi*hi + hi*lo + lo*hi (lo*lo ~2^-16, negligible)
      acc[t] = __builtin_amdgcn_wmma_f32_16x16x32_bf16(false, ahi, false, bhi,
                                                       (short)0, acc[t], false, false);
      acc[t] = __builtin_amdgcn_wmma_f32_16x16x32_bf16(false, ahi, false, blo,
                                                       (short)0, acc[t], false, false);
      acc[t] = __builtin_amdgcn_wmma_f32_16x16x32_bf16(false, alo, false, bhi,
                                                       (short)0, acc[t], false, false);
    }
  }

  // ---- store C: f32 16x16 layout (VGPR i -> M = i + 8*half, N = l16) ----
  float* yp = Y + (size_t)(row0 + wave * 16 + 8 * half) * N + (n0 + l16);
#pragma unroll
  for (int i = 0; i < 8; ++i) {
#pragma unroll
    for (int t = 0; t < NT; ++t) yp[(size_t)i * N + t * 16] = acc[t][i];
  }
}

// ---------------------------------------------------------------------------
// Stage 2: rank contraction + RoPE.  grid = 16384 blocks, 64 threads.
// Thread p owns the rotary pair (d=p, d=p+64) so no cross-lane exchange.
// ---------------------------------------------------------------------------
__global__ __launch_bounds__(64)
void cpl_combine_rope(const float* __restrict__ Aq, const float* __restrict__ Ak,
                      const float* __restrict__ Av, const float* __restrict__ Bq,
                      const float* __restrict__ Bk, const float* __restrict__ Bv,
                      float* __restrict__ out) {
  const int bs = blockIdx.x;          // flattened (b, s)
  const int s  = bs & (SEQ - 1);      // position within sequence
  const int p  = threadIdx.x;         // 0..63 (rotary pair index)

  // inv_freq = 10000^(-p/64); angle = s * inv_freq  (reference math, f32)
  const float inv = expf((float)p * -0.14391156816f);   // -ln(1e4)/64
  float sn, cs;
  sincosf((float)s * inv, &sn, &cs);

  const float* aq = Aq + (size_t)bs * (NHEAD * QRANK);
  const float* ak = Ak + (size_t)bs * (NHEAD * KVRANK);
  const float* av = Av + (size_t)bs * (NHEAD * KVRANK);
  const float* bq = Bq + (size_t)bs * (QRANK * HDIM);
  const float* bk = Bk + (size_t)bs * (KVRANK * HDIM);
  const float* bv = Bv + (size_t)bs * (KVRANK * HDIM);

  constexpr size_t OUTSZ = (size_t)BSROWS * NHEAD * HDIM;
  float* qo = out;
  float* ko = out + OUTSZ;
  float* vo = out + 2 * OUTSZ;
  const size_t rowBase = (size_t)bs * NHEAD * HDIM;

  // ---- q: rank 8 ----
  float bq1[QRANK], bq2[QRANK];
#pragma unroll
  for (int r = 0; r < QRANK; ++r) {
    bq1[r] = bq[r * HDIM + p];
    bq2[r] = bq[r * HDIM + 64 + p];
  }
#pragma unroll
  for (int h = 0; h < NHEAD; ++h) {
    float a1 = 0.f, a2 = 0.f;
#pragma unroll
    for (int r = 0; r < QRANK; ++r) {
      float a = aq[h * QRANK + r];          // wave-uniform -> scalar load
      a1 += a * bq1[r];
      a2 += a * bq2[r];
    }
    a1 *= 0.125f; a2 *= 0.125f;             // / Q_RANK
    qo[rowBase + (size_t)h * HDIM + p]      =  a1 * cs + a2 * sn;
    qo[rowBase + (size_t)h * HDIM + 64 + p] = -a1 * sn + a2 * cs;
  }

  // ---- k: rank 2, with RoPE ----
  float bk1[KVRANK], bk2[KVRANK];
#pragma unroll
  for (int r = 0; r < KVRANK; ++r) {
    bk1[r] = bk[r * HDIM + p];
    bk2[r] = bk[r * HDIM + 64 + p];
  }
#pragma unroll
  for (int h = 0; h < NHEAD; ++h) {
    float c0 = ak[h * KVRANK + 0], c1 = ak[h * KVRANK + 1];
    float a1 = 0.5f * (c0 * bk1[0] + c1 * bk1[1]);
    float a2 = 0.5f * (c0 * bk2[0] + c1 * bk2[1]);
    ko[rowBase + (size_t)h * HDIM + p]      =  a1 * cs + a2 * sn;
    ko[rowBase + (size_t)h * HDIM + 64 + p] = -a1 * sn + a2 * cs;
  }

  // ---- v: rank 2, no RoPE ----
  float bv1[KVRANK], bv2[KVRANK];
#pragma unroll
  for (int r = 0; r < KVRANK; ++r) {
    bv1[r] = bv[r * HDIM + p];
    bv2[r] = bv[r * HDIM + 64 + p];
  }
#pragma unroll
  for (int h = 0; h < NHEAD; ++h) {
    float c0 = av[h * KVRANK + 0], c1 = av[h * KVRANK + 1];
    vo[rowBase + (size_t)h * HDIM + p]      = 0.5f * (c0 * bv1[0] + c1 * bv1[1]);
    vo[rowBase + (size_t)h * HDIM + 64 + p] = 0.5f * (c0 * bv2[0] + c1 * bv2[1]);
  }
}

// ---------------------------------------------------------------------------
extern "C" void kernel_launch(void* const* d_in, const int* in_sizes, int n_in,
                              void* d_out, int out_size, void* d_ws, size_t ws_size,
                              hipStream_t stream) {
  (void)in_sizes; (void)n_in; (void)out_size; (void)ws_size;
  const float* x    = (const float*)d_in[0];
  const float* Wa_q = (const float*)d_in[1];
  const float* Wa_k = (const float*)d_in[2];
  const float* Wa_v = (const float*)d_in[3];
  const float* Wb_q = (const float*)d_in[4];
  const float* Wb_k = (const float*)d_in[5];
  const float* Wb_v = (const float*)d_in[6];
  float* out = (float*)d_out;

  // ---- workspace layout ------------------------------------------------
  // [f32 Y regions 113 MB][bf16 Xhi 67 MB][bf16 Xlo 67 MB][bf16 Wt hi/lo 14 MB]
  const size_t R = (size_t)BSROWS;
  float* wsf = (float*)d_ws;
  float* Yaq = wsf;
  float* Yak = Yaq + R * 128;
  float* Yav = Yak + R * 32;
  float* Ybq = Yav + R * 32;
  float* Ybk = Ybq + R * 1024;
  float* Ybv = Ybk + R * 256;
  __bf16* bws = (__bf16*)(Ybv + R * 256);
  const size_t XN = R * D_IN;                 // 33.5M elements
  __bf16* Xhi = bws;          __bf16* Xlo = Xhi + XN;
  __bf16* Wt  = Xlo + XN;     // per-weight transposed hi/lo blocks
  __bf16* WaqH = Wt;                 __bf16* WaqL = WaqH + (size_t)128  * D_IN;
  __bf16* WakH = WaqL + (size_t)128 * D_IN;  __bf16* WakL = WakH + (size_t)32  * D_IN;
  __bf16* WavH = WakL + (size_t)32  * D_IN;  __bf16* WavL = WavH + (size_t)32  * D_IN;
  __bf16* WbqH = WavL + (size_t)32  * D_IN;  __bf16* WbqL = WbqH + (size_t)1024 * D_IN;
  __bf16* WbkH = WbqL + (size_t)1024* D_IN;  __bf16* WbkL = WbkH + (size_t)256 * D_IN;
  __bf16* WbvH = WbkL + (size_t)256 * D_IN;  __bf16* WbvL = WbvH + (size_t)256 * D_IN;

  // ---- stage 0: split / transpose -------------------------------------
  {
    int n4 = (int)(XN / 4);
    cpl_split<<<n4 / 256, 256, 0, stream>>>(x, Xhi, Xlo, n4);
    cpl_split_transpose<<<(128  * D_IN) / 256, 256, 0, stream>>>(Wa_q, WaqH, WaqL, 128);
    cpl_split_transpose<<<(32   * D_IN) / 256, 256, 0, stream>>>(Wa_k, WakH, WakL, 32);
    cpl_split_transpose<<<(32   * D_IN) / 256, 256, 0, stream>>>(Wa_v, WavH, WavL, 32);
    cpl_split_transpose<<<(1024 * D_IN) / 256, 256, 0, stream>>>(Wb_q, WbqH, WbqL, 1024);
    cpl_split_transpose<<<(256  * D_IN) / 256, 256, 0, stream>>>(Wb_k, WbkH, WbkL, 256);
    cpl_split_transpose<<<(256  * D_IN) / 256, 256, 0, stream>>>(Wb_v, WbvH, WbvL, 256);
  }

  // ---- stage 1: WMMA GEMMs --------------------------------------------
  const dim3 blk(256);
  const int mg = BSROWS / TILE_M;   // 128
  cpl_gemm_wmma<4><<<dim3(mg, 128  / 64), blk, 0, stream>>>(Xhi, Xlo, WaqH, WaqL, Yaq, 128);
  cpl_gemm_wmma<2><<<dim3(mg, 32   / 32), blk, 0, stream>>>(Xhi, Xlo, WakH, WakL, Yak, 32);
  cpl_gemm_wmma<2><<<dim3(mg, 32   / 32), blk, 0, stream>>>(Xhi, Xlo, WavH, WavL, Yav, 32);
  cpl_gemm_wmma<4><<<dim3(mg, 1024 / 64), blk, 0, stream>>>(Xhi, Xlo, WbqH, WbqL, Ybq, 1024);
  cpl_gemm_wmma<4><<<dim3(mg, 256  / 64), blk, 0, stream>>>(Xhi, Xlo, WbkH, WbkL, Ybk, 256);
  cpl_gemm_wmma<4><<<dim3(mg, 256  / 64), blk, 0, stream>>>(Xhi, Xlo, WbvH, WbvL, Ybv, 256);

  // ---- stage 2: combine + RoPE ----------------------------------------
  cpl_combine_rope<<<BSROWS, 64, 0, stream>>>(Yaq, Yak, Yav, Ybq, Ybk, Ybv, out);
}
